// MultiTaskModel_7241314861674
// MI455X (gfx1250) — matrix operand order repaired and checked by reading, not compiled, not run on hardware
//
#include <hip/hip_runtime.h>
#include <hip/hip_bf16.h>

// ---------------------------------------------------------------------------
// MI455X / gfx1250 implementation of the MoE segmentation model.
//
// f16 NHWC activations + per-launch weight transpose [O,C,KH,KW] ->
// [O,KH*KW,C] f16 make the implicit-GEMM K dimension contiguous for both
// operands. Non-stem layers have C % 32 == 0, so a 32-wide K chunk lies in
// one (ky,kx) tap and one concat source: bounds / upsample / concat select
// hoist out of the per-element path; fragments are pure 16-byte vector
// loads (v8h == global_load_b128).
//
// Each wave owns a 64(M) x 32(N) output tile: 4 A frags x 2 B frags ->
// 8 V_WMMA_F32_16X16X32_F16 per K chunk against 12 b128 loads; B chunk for
// the next step is prefetched with global_prefetch_b8.  Accumulate f32;
// BN/residual/ReLU epilogue in f32, stored f16 (final seg stored f32 NCHW).
// Fused: SAME padding, stride 1/2, folded BN, residual, ReLU, nearest 2x
// upsample + concat, per-sample MoE expert select (64-row M tiles never
// straddle a sample: Hout*Wout % 64 == 0 for every layer).
// The 7x7 stem (C=3, Ktot=147) keeps a dedicated constexpr gather kernel.
//
// WMMA fragment mappings (CDNA5 ISA, wave32):
//   A (16x32 f16, MxK): M = lane%16 ; elem j -> K = hi*8 + (j<8 ? j : 16+j-8)
//   B (32x16 f16, KxN): N = lane%16 ; elem j -> K = hi*8 + (j<8 ? j : 16+j-8)
//   C/D (16x16 f32):    N = lane%16 ; vgpr i -> M = i + 8*hi   (hi = lane>=16)
// ---------------------------------------------------------------------------

typedef __attribute__((ext_vector_type(16))) _Float16 v16h;
typedef __attribute__((ext_vector_type(8)))  _Float16 v8h;
typedef __attribute__((ext_vector_type(8)))  float    v8f;
typedef _Float16 half_t;

union Frag16 { v16h v; v8h q[2]; _Float16 h[16]; };

#define WMMA_F16(A, B, C) __builtin_amdgcn_wmma_f32_16x16x32_f16( \
    false, (A), false, (B), (short)0, (C), false, false)

// ======================= fast path: NHWC f16, C % 32 == 0 =======================
template <int KH, int KW, int UPS>
__global__ __launch_bounds__(256)
void conv_wmma_nhwc(
    const half_t* __restrict__ in0, int C0,           // NHWC; UPS -> stored H/2 x W/2
    const half_t* __restrict__ in1, int C1,           // NHWC concat tail (may be null)
    const half_t* __restrict__ wgt,                   // [E?][Cout][KH*KW][C] f16
    const float* __restrict__ scale,                  // [E?][Cout] or null (=1)
    const float* __restrict__ bias,                   // [E?][Cout] or null (=0)
    const half_t* __restrict__ res,                   // residual, NHWC f16, or null
    half_t* __restrict__ out_h,                       // NHWC f16 (when !out_nchw)
    float* __restrict__ out_f,                        // NCHW f32 (when out_nchw)
    int N, int H, int W,                              // logical input dims (post-upsample)
    int Cout, int stride,
    int Hout, int Wout, int relu, int out_nchw,
    const int* __restrict__ sel, long wStrideE, int sbStrideE)
{
    constexpr int KHW = KH * KW;

    const int lane = threadIdx.x & 31;
    const int wave = threadIdx.x >> 5;
    const long HW     = (long)Hout * Wout;            // multiple of 64 for all layers
    const long tilesM = (long)N * HW / 64;
    const int  tilesN = (Cout + 31) >> 5;
    const long tile   = (long)blockIdx.x * 8 + wave;
    if (tile >= tilesM * (long)tilesN) return;        // wave-uniform exit

    const int  tn = (int)(tile / tilesM);
    const long tm = tile % tilesM;
    const int  l15 = lane & 15;
    const int  hi  = lane >> 4;

    const long m0    = tm * 64 + l15;
    const int  n_img = (int)(m0 / HW);                // uniform across the wave

    int pt = (Hout - 1) * stride + KH - H; const int padY = pt > 0 ? (pt >> 1) : 0;
    pt     = (Wout - 1) * stride + KW - W; const int padX = pt > 0 ? (pt >> 1) : 0;

    // four M sub-tiles (rows m0 + 16r), all inside one sample
    int iyR[4], ixR[4];
#pragma unroll
    for (int r = 0; r < 4; ++r) {
        const int rem = (int)((m0 + 16 * r) % HW);
        const int oy  = rem / Wout, ox = rem - oy * Wout;
        iyR[r] = oy * stride - padY;
        ixR[r] = ox * stride - padX;
    }

    const int C    = C0 + C1;                         // multiple of 32 (caller guarantees)
    const int Ktot = C * KHW;
    const int ex   = sel ? sel[n_img] : 0;

    const int nc0 = tn * 32 + l15;
    const int nc1 = nc0 + 16;
    const half_t* wbase = wgt + (long)ex * wStrideE;
    const half_t* wrow0 = (nc0 < Cout) ? wbase + (long)nc0 * Ktot + hi * 8 : nullptr;
    const half_t* wrow1 = (nc1 < Cout) ? wbase + (long)nc1 * Ktot + hi * 8 : nullptr;

    const int H0 = UPS ? (H >> 1) : H;
    const int W0 = UPS ? (W >> 1) : W;
    const half_t* in0n = in0 + (long)n_img * H0 * W0 * C0;
    const half_t* in1n = in1 ? (in1 + (long)n_img * (long)H * W * C1) : nullptr;

    const v8h zero = {};
    v8f acc[4][2] = {};

    int ky = 0, kx = 0, k = 0;                        // k advances by 32 per chunk
    for (int s = 0; s < KHW; ++s) {
        // per-tap pixel validity + offsets (hoisted out of the channel loop)
        bool vR[4];
        long off0[4], off1[4];
#pragma unroll
        for (int r = 0; r < 4; ++r) {
            const int iy = iyR[r] + ky, ix = ixR[r] + kx;
            vR[r]   = (unsigned)iy < (unsigned)H && (unsigned)ix < (unsigned)W;
            off0[r] = ((long)(UPS ? (iy >> 1) : iy) * W0 + (UPS ? (ix >> 1) : ix)) * C0;
            off1[r] = ((long)iy * W + ix) * C1;
        }

        for (int cb = 0; cb < C; cb += 32, k += 32) {
            // chunk never crosses the concat boundary (C0 % 32 == 0)
            const bool first = cb < C0;               // wave-uniform
            Frag16 a[4], b0, b1;
#pragma unroll
            for (int r = 0; r < 4; ++r) {
                const half_t* p = (first ? in0n + off0[r] + cb
                                         : in1n + off1[r] + (cb - C0)) + hi * 8;
                if (vR[r]) {                           // two 16-byte vector loads
                    a[r].q[0] = *(const v8h*)(p);
                    a[r].q[1] = *(const v8h*)(p + 16);
                } else { a[r].q[0] = zero; a[r].q[1] = zero; }
            }
            if (wrow0) {
                b0.q[0] = *(const v8h*)(wrow0 + k);
                b0.q[1] = *(const v8h*)(wrow0 + k + 16);
                __builtin_prefetch(wrow0 + k + 32, 0, 1);   // global_prefetch_b8
            } else { b0.q[0] = zero; b0.q[1] = zero; }
            if (wrow1) {
                b1.q[0] = *(const v8h*)(wrow1 + k);
                b1.q[1] = *(const v8h*)(wrow1 + k + 16);
                __builtin_prefetch(wrow1 + k + 32, 0, 1);
            } else { b1.q[0] = zero; b1.q[1] = zero; }
#pragma unroll
            for (int r = 0; r < 4; ++r) {
                acc[r][0] = WMMA_F16(a[r].v, b0.v, acc[r][0]);
                acc[r][1] = WMMA_F16(a[r].v, b1.v, acc[r][1]);
            }
        }
        if (++kx == KW) { kx = 0; ++ky; }
    }

    auto store_tile = [&](const v8f& a, int ma, int nc) {
        if (nc >= Cout) return;
        const float sc = scale ? scale[(long)ex * sbStrideE + nc] : 1.0f;
        const float bb = bias  ? bias [(long)ex * sbStrideE + nc] : 0.0f;
#pragma unroll
        for (int i = 0; i < 8; ++i) {
            const long row = tm * 64 + ma * 16 + i + hi * 8;
            const int  rr  = (int)(row % HW);
            const int  ry  = rr / Wout;
            const int  rx  = rr - ry * Wout;
            float v = a[i] * sc + bb;
            if (out_nchw) {
                out_f[(((long)n_img * Cout + nc) * Hout + ry) * Wout + rx] = v;
            } else {
                const long oidx = (((long)n_img * Hout + ry) * Wout + rx) * Cout + nc;
                if (res)  v += (float)res[oidx];
                if (relu) v = v > 0.0f ? v : 0.0f;
                out_h[oidx] = (half_t)v;
            }
        }
    };
#pragma unroll
    for (int r = 0; r < 4; ++r) {
        store_tile(acc[r][0], r, nc0);
        store_tile(acc[r][1], r, nc1);
    }
}

// ===================== stem: 7x7 s2, C=3, fully hardcoded =====================
__global__ __launch_bounds__(256)
void conv_stem_kernel(const half_t* __restrict__ in,   // NHWC f16 [4,512,512,3]
                      const half_t* __restrict__ wgt,  // [64,49,3] f16 (transposed)
                      const float* __restrict__ scale,
                      const float* __restrict__ bias,
                      half_t* __restrict__ out)        // NHWC f16 [4,256,256,64]
{
    constexpr int H = 512, W = 512, Ho = 256, Wo = 256;
    constexpr int CIN = 3, Cout = 64, Ktot = 147, PAD = 2;
    const int lane = threadIdx.x & 31;
    const int wave = threadIdx.x >> 5;
    const long HW     = (long)Ho * Wo;
    const long tilesM = 4 * HW / 32;
    const long tile   = (long)blockIdx.x * 8 + wave;
    if (tile >= tilesM * 2) return;
    const int  tn = (int)(tile / tilesM);
    const long tm = tile % tilesM;
    const int  l15 = lane & 15, hi = lane >> 4;

    const long m0    = tm * 32 + l15;
    const int  n_img = (int)(m0 / HW);
    const int  rem0  = (int)(m0 % HW);
    const int  rem1  = (int)((m0 + 16) % HW);
    const int  oy0 = rem0 / Wo, ox0 = rem0 - oy0 * Wo;
    const int  oy1 = rem1 / Wo, ox1 = rem1 - oy1 * Wo;
    const int  iyA = oy0 * 2 - PAD, ixA = ox0 * 2 - PAD;
    const int  iyB = oy1 * 2 - PAD, ixB = ox1 * 2 - PAD;

    const int nc0 = tn * 32 + l15, nc1 = nc0 + 16;
    const half_t* wrow0 = wgt + nc0 * Ktot;
    const half_t* wrow1 = wgt + nc1 * Ktot;
    const half_t* inn   = in + (long)n_img * H * W * CIN;
    const int kAoff = hi * 8, kBoff = hi * 8;
    const _Float16 hz = (_Float16)0.0f;

    v8f acc00 = {}, acc01 = {}, acc10 = {}, acc11 = {};
    for (int k0 = 0; k0 < Ktot; k0 += 32) {
        Frag16 a0, a1, b0, b1;
#pragma unroll
        for (int t = 0; t < 16; ++t) {
            const int ka = k0 + ((t < 8) ? t : t + 8) + kAoff;
            _Float16 av0 = hz, av1 = hz;
            if (ka < Ktot) {                          // constant divisors 3 / 7
                const int s  = ka / 3, c = ka - s * 3;
                const int ky = s / 7,  kx = s - ky * 7;
                int iy = iyA + ky, ix = ixA + kx;
                if ((unsigned)iy < (unsigned)H && (unsigned)ix < (unsigned)W)
                    av0 = inn[((long)iy * W + ix) * CIN + c];
                iy = iyB + ky; ix = ixB + kx;
                if ((unsigned)iy < (unsigned)H && (unsigned)ix < (unsigned)W)
                    av1 = inn[((long)iy * W + ix) * CIN + c];
            }
            a0.h[t] = av0;
            a1.h[t] = av1;
            const int kb  = k0 + ((t < 8) ? t : t + 8) + kBoff;
            const bool ki = kb < Ktot;
            b0.h[t] = ki ? wrow0[kb] : hz;
            b1.h[t] = ki ? wrow1[kb] : hz;
        }
        acc00 = WMMA_F16(a0.v, b0.v, acc00);
        acc01 = WMMA_F16(a0.v, b1.v, acc01);
        acc10 = WMMA_F16(a1.v, b0.v, acc10);
        acc11 = WMMA_F16(a1.v, b1.v, acc11);
    }
    auto st = [&](const v8f& acc, int ma, int nc) {
        const float sc = scale[nc], bb = bias[nc];
#pragma unroll
        for (int i = 0; i < 8; ++i) {
            const long row = tm * 32 + ma * 16 + i + hi * 8;
            const int  rr  = (int)(row % HW);
            const int  ry  = rr / Wo, rx = rr - ry * Wo;
            float v = acc[i] * sc + bb;
            v = v > 0.0f ? v : 0.0f;
            out[(((long)n_img * Ho + ry) * Wo + rx) * Cout + nc] = (half_t)v;
        }
    };
    st(acc00, 0, nc0); st(acc01, 0, nc1); st(acc10, 1, nc0); st(acc11, 1, nc1);
}

// ======================= layout / pooling / classifier =======================
// f32 NCHW -> f16 NHWC
__global__ void nchw_to_nhwc_kernel(const float* __restrict__ src, half_t* __restrict__ dst,
                                    int N, int C, int H, int W)
{
    long i = (long)blockIdx.x * blockDim.x + threadIdx.x;       // dst index, c fastest
    const long total = (long)N * C * H * W;
    if (i >= total) return;
    const int c = (int)(i % C); long r = i / C;
    const int w = (int)(r % W); r /= W;
    const int h = (int)(r % H);
    const int n = (int)(r / H);
    dst[i] = (half_t)src[(((long)n * C + c) * H + h) * W + w];
}

// [Ot, C, KHW] f32 -> [Ot, KHW, C] f16   (Ot folds any expert dim; KHW=1 => convert)
__global__ void transpose_w_kernel(const float* __restrict__ src, half_t* __restrict__ dst,
                                   int Ot, int C, int KHW)
{
    long i = (long)blockIdx.x * blockDim.x + threadIdx.x;       // dst index, c fastest
    const long total = (long)Ot * C * KHW;
    if (i >= total) return;
    const int c = (int)(i % C); long r = i / C;
    const int s = (int)(r % KHW);
    const int o = (int)(r / KHW);
    dst[i] = (half_t)src[((long)o * C + c) * KHW + s];
}

// 3x3 stride-2 SAME maxpool, NHWC f16
__global__ void maxpool_nhwc_kernel(const half_t* __restrict__ in, half_t* __restrict__ out,
                                    int N, int C, int H, int W, int Hout, int Wout)
{
    long i = (long)blockIdx.x * blockDim.x + threadIdx.x;
    const long total = (long)N * Hout * Wout * C;
    if (i >= total) return;
    const int c  = (int)(i % C); long r = i / C;
    const int ox = (int)(r % Wout); r /= Wout;
    const int oy = (int)(r % Hout);
    const int n  = (int)(r / Hout);
    int pt = (Hout - 1) * 2 + 3 - H; const int pad = pt > 0 ? (pt >> 1) : 0;
    const int iy0 = oy * 2 - pad, ix0 = ox * 2 - pad;
    const half_t* p = in + (long)n * H * W * C + c;
    float m = -__builtin_inff();
    for (int ky = 0; ky < 3; ++ky) {
        const int iy = iy0 + ky;
        if ((unsigned)iy >= (unsigned)H) continue;
        for (int kx = 0; kx < 3; ++kx) {
            const int ix = ix0 + kx;
            if ((unsigned)ix >= (unsigned)W) continue;
            const float v = (float)p[((long)iy * W + ix) * C];
            m = v > m ? v : m;
        }
    }
    out[i] = (half_t)m;
}

// NHWC f16 global average pool -> f32 [N,C]
__global__ void avgpool_nhwc_kernel(const half_t* __restrict__ in, float* __restrict__ out,
                                    int N, int C, int HW)
{
    const int i = blockIdx.x * blockDim.x + threadIdx.x;
    if (i >= N * C) return;
    const int n = i / C, c = i - n * C;
    const half_t* p = in + (long)n * HW * C + c;
    float s = 0.0f;
    for (int k = 0; k < HW; ++k) s += (float)p[(long)k * C];
    out[i] = s / (float)HW;
}

// pooled[4,512] f32 -> relu(fc 512->128) -> fc 128->5 -> logits + argmax
__global__ void classifier_kernel(const float* __restrict__ pooled,
                                  const float* __restrict__ w1, const float* __restrict__ b1,
                                  const float* __restrict__ w2, const float* __restrict__ b2,
                                  float* __restrict__ out, int* __restrict__ sel)
{
    __shared__ float hc[4][128];
    __shared__ float lg[4][5];
    const int t = threadIdx.x;                     // 128 threads
    for (int n = 0; n < 4; ++n) {
        float s = b1[t];
        const float* p  = pooled + n * 512;
        const float* wr = w1 + t * 512;
        for (int k = 0; k < 512; ++k) s += p[k] * wr[k];
        hc[n][t] = s > 0.0f ? s : 0.0f;
    }
    __syncthreads();
    if (t < 20) {
        const int n = t / 5, j = t % 5;
        float s = b2[j];
        const float* wr = w2 + j * 128;
        for (int k = 0; k < 128; ++k) s += hc[n][k] * wr[k];
        lg[n][j]       = s;
        out[n * 5 + j] = s;                         // logits -> d_out[0..19]
    }
    __syncthreads();
    if (t < 4) {                                    // first-max argmax (jnp.argmax)
        int best = 0; float bv = lg[t][0];
        for (int j = 1; j < 5; ++j) if (lg[t][j] > bv) { bv = lg[t][j]; best = j; }
        sel[t]      = best;
        out[20 + t] = (float)best;                  // sel -> d_out[20..23]
    }
}

// ============================== host orchestration ==============================
static inline void launch_conv(hipStream_t stream,
    const half_t* in0, int C0, int ups, const half_t* in1, int C1,
    const half_t* w /* transposed f16 */, const float* s, const float* b,
    const half_t* res, half_t* out_h, float* out_f32,
    int N, int H, int W, int Cout, int KH, int stride,
    int relu, int out_nchw, const int* sel, long wStrideE, int sbStrideE)
{
    const int Hout = (H + stride - 1) / stride;
    const int Wout = (W + stride - 1) / stride;
    const long tiles = ((long)N * Hout * Wout / 64) * ((Cout + 31) / 32);
    const int blocks = (int)((tiles + 7) / 8);
#define LCONV(KKH, KKW, UU)                                                     \
    conv_wmma_nhwc<KKH, KKW, UU><<<blocks, 256, 0, stream>>>(                   \
        in0, C0, in1, C1, w, s, b, res, out_h, out_f32, N, H, W, Cout, stride,  \
        Hout, Wout, relu, out_nchw, sel, wStrideE, sbStrideE)
    if (KH == 3 && !ups)      { LCONV(3, 3, 0); }
    else if (KH == 3 &&  ups) { LCONV(3, 3, 1); }
    else if (KH == 1 && !ups) { LCONV(1, 1, 0); }
    else                      { LCONV(1, 1, 1); }
#undef LCONV
}

extern "C" void kernel_launch(void* const* d_in, const int* in_sizes, int n_in,
                              void* d_out, int out_size, void* d_ws, size_t ws_size,
                              hipStream_t stream)
{
    (void)in_sizes; (void)n_in; (void)out_size; (void)ws_size;

    // -------- input pointer map (jax tree flatten: dict keys sorted, lists in order)
    const float* cls_b1 = (const float*)d_in[0];
    const float* cls_b2 = (const float*)d_in[1];
    const float* cls_w1 = (const float*)d_in[2];
    const float* cls_w2 = (const float*)d_in[3];
    const float* enc_b0 = (const float*)d_in[4];
    // non-down block: b1,b2,s1,s2,w1,w2 (6) ; down block: b1,b2,bd,s1,s2,sd,w1,w2,wd (9)
    // l1b1 @5, l1b2 @11, l2b1 @17(d), l2b2 @26, l3b1 @32(d), l3b2 @41, l4b1 @47(d), l4b2 @56
    const float* enc_s0 = (const float*)d_in[62];
    const float* enc_w0 = (const float*)d_in[63];
    // params.experts: bout @64, stages[0..3] (b1,b2,s1,s2,w1,w2) @65,71,77,83, wout @89
    const float* ex_bout = (const float*)d_in[64];
    const float* ex_wout = (const float*)d_in[89];
    const float* x       = (const float*)d_in[90];

    // -------- workspace: byte bump allocator (256-B aligned blocks)
    char* base = (char*)d_ws;
    size_t off = 0;
    auto AB = [&](size_t bytes) -> void* {
        void* p = base + off; off = (off + bytes + 255) & ~(size_t)255; return p;
    };
    auto AH = [&](size_t n) -> half_t* { return (half_t*)AB(n * sizeof(half_t)); };

    half_t* xh    = AH(3145728);    // x NHWC f16 [4,512,512,3]
    half_t* e1buf = AH(16777216);   // [4,256,256,64]
    half_t* pbuf  = AH(4194304);    // [4,128,128,64]
    half_t* tbuf  = AH(16777216);   // block/stage intermediates (max [4,256,256,64])
    half_t* scbuf = AH(2097152);    // downsample shortcuts
    half_t* l1a   = AH(4194304);
    half_t* e2buf = AH(4194304);
    half_t* l2a   = AH(2097152);
    half_t* e3buf = AH(2097152);
    half_t* l3a   = AH(1048576);
    half_t* e4buf = AH(1048576);
    half_t* l4a   = AH(524288);
    half_t* e5buf = AH(524288);
    half_t* d1buf = AH(1048576);
    half_t* d2buf = AH(2097152);
    half_t* d3buf = AH(4194304);
    half_t* d4buf = AH(16777216);
    float*  pooled= (float*)AB(2048 * sizeof(float));
    int*    selbuf= (int*)AB(64);

    // transpose+convert helper: [Ot,C,KHW] f32 -> [Ot,KHW,C] f16
    auto tw = [&](const float* src, int Ot, int C, int KHW) -> const half_t* {
        const long total = (long)Ot * C * KHW;
        half_t* dst = AH((size_t)total);
        transpose_w_kernel<<<(int)((total + 255) / 256), 256, 0, stream>>>(src, dst, Ot, C, KHW);
        return dst;
    };

    float* out_f = (float*)d_out;        // [0..19] logits, [20..23] sel, [24..] seg NCHW f32

    // -------- layout conversions / weight transposes
    {
        const long total = 4L * 3 * 512 * 512;
        nchw_to_nhwc_kernel<<<(int)((total + 255) / 256), 256, 0, stream>>>(x, xh, 4, 3, 512, 512);
    }
    const half_t* w0T = tw(enc_w0, 64, 3, 49);

    // ================= encoder =================
    {   // stem: 7x7 s2 conv + BN + ReLU -> e1 [4,256,256,64] NHWC
        const long tiles = (4L * 256 * 256 / 32) * 2;
        conv_stem_kernel<<<(int)((tiles + 7) / 8), 256, 0, stream>>>(xh, w0T, enc_s0, enc_b0, e1buf);
    }
    {   // 3x3 s2 maxpool -> [4,128,128,64]
        const long total = 4L * 128 * 128 * 64;
        maxpool_nhwc_kernel<<<(int)((total + 255) / 256), 256, 0, stream>>>(
            e1buf, pbuf, 4, 64, 256, 256, 128, 128);
    }

    // basic block (identity shortcut)
    auto block = [&](const half_t* in, half_t* outp, int bidx, int Cio, int Hh) {
        const float* b1 = (const float*)d_in[bidx + 0];
        const float* b2 = (const float*)d_in[bidx + 1];
        const float* s1 = (const float*)d_in[bidx + 2];
        const float* s2 = (const float*)d_in[bidx + 3];
        const half_t* w1T = tw((const float*)d_in[bidx + 4], Cio, Cio, 9);
        const half_t* w2T = tw((const float*)d_in[bidx + 5], Cio, Cio, 9);
        launch_conv(stream, in, Cio, 0, nullptr, 0, w1T, s1, b1, nullptr,
                    tbuf, nullptr, 4, Hh, Hh, Cio, 3, 1, 1, 0, nullptr, 0, 0);
        launch_conv(stream, tbuf, Cio, 0, nullptr, 0, w2T, s2, b2, in,
                    outp, nullptr, 4, Hh, Hh, Cio, 3, 1, 1, 0, nullptr, 0, 0);
    };
    // down block (1x1 s2 conv+BN shortcut)
    auto dblock = [&](const half_t* in, half_t* outp, int bidx, int Cin, int Co, int Hh) {
        const float* b1 = (const float*)d_in[bidx + 0];
        const float* b2 = (const float*)d_in[bidx + 1];
        const float* bd = (const float*)d_in[bidx + 2];
        const float* s1 = (const float*)d_in[bidx + 3];
        const float* s2 = (const float*)d_in[bidx + 4];
        const float* sd = (const float*)d_in[bidx + 5];
        const half_t* w1T = tw((const float*)d_in[bidx + 6], Co, Cin, 9);
        const half_t* w2T = tw((const float*)d_in[bidx + 7], Co, Co, 9);
        const half_t* wdT = tw((const float*)d_in[bidx + 8], Co, Cin, 1);  // f16 convert
        const int Ho = Hh / 2;
        launch_conv(stream, in, Cin, 0, nullptr, 0, w1T, s1, b1, nullptr,
                    tbuf, nullptr, 4, Hh, Hh, Co, 3, 2, 1, 0, nullptr, 0, 0);
        launch_conv(stream, in, Cin, 0, nullptr, 0, wdT, sd, bd, nullptr,
                    scbuf, nullptr, 4, Hh, Hh, Co, 1, 2, 0, 0, nullptr, 0, 0);
        launch_conv(stream, tbuf, Co, 0, nullptr, 0, w2T, s2, b2, scbuf,
                    outp, nullptr, 4, Ho, Ho, Co, 3, 1, 1, 0, nullptr, 0, 0);
    };

    block (pbuf,  l1a,   5,  64, 128);        // l1b1
    block (l1a,   e2buf, 11, 64, 128);        // l1b2 -> e2 [4,128,128,64]
    dblock(e2buf, l2a,   17, 64, 128, 128);   // l2b1 -> [4,64,64,128]
    block (l2a,   e3buf, 26, 128, 64);        // l2b2 -> e3
    dblock(e3buf, l3a,   32, 128, 256, 64);   // l3b1 -> [4,32,32,256]
    block (l3a,   e4buf, 41, 256, 32);        // l3b2 -> e4
    dblock(e4buf, l4a,   47, 256, 512, 32);   // l4b1 -> [4,16,16,512]
    block (l4a,   e5buf, 56, 512, 16);        // l4b2 -> e5

    // ================= classifier + expert selection =================
    avgpool_nhwc_kernel<<<(4 * 512 + 255) / 256, 256, 0, stream>>>(e5buf, pooled, 4, 512, 256);
    classifier_kernel<<<1, 128, 0, stream>>>(pooled, cls_w1, cls_b1, cls_w2, cls_b2,
                                             out_f, selbuf);

    // ================= MoE decoder (per-sample expert via selbuf) =================
    auto stage = [&](const half_t* xin, int Cx, const half_t* skip, int Cs,
                     half_t* outp, int bidx, int Co, int Hh /* post-upsample */) {
        const float* b1 = (const float*)d_in[bidx + 0];
        const float* b2 = (const float*)d_in[bidx + 1];
        const float* s1 = (const float*)d_in[bidx + 2];
        const float* s2 = (const float*)d_in[bidx + 3];
        const half_t* w1T = tw((const float*)d_in[bidx + 4], 5 * Co, Cx + Cs, 9);
        const half_t* w2T = tw((const float*)d_in[bidx + 5], 5 * Co, Co, 9);
        const long w1E = (long)Co * (Cx + Cs) * 9;
        const long w2E = (long)Co * Co * 9;
        launch_conv(stream, xin, Cx, 1, skip, Cs, w1T, s1, b1, nullptr,
                    tbuf, nullptr, 4, Hh, Hh, Co, 3, 1, 1, 0, selbuf, w1E, Co);
        launch_conv(stream, tbuf, Co, 0, nullptr, 0, w2T, s2, b2, nullptr,
                    outp, nullptr, 4, Hh, Hh, Co, 3, 1, 1, 0, selbuf, w2E, Co);
    };
    stage(e5buf, 512, e4buf, 256, d1buf, 65, 256, 32);    // -> [4,32,32,256]
    stage(d1buf, 256, e3buf, 128, d2buf, 71, 128, 64);    // -> [4,64,64,128]
    stage(d2buf, 128, e2buf, 64,  d3buf, 77, 64, 128);    // -> [4,128,128,64]
    stage(d3buf, 64,  e1buf, 64,  d4buf, 83, 64, 256);    // -> [4,256,256,64]

    // final: up2 to 512, per-expert 1x1 conv + bias -> seg [4,5,512,512] (NCHW f32)
    const half_t* woutT = tw(ex_wout, 25, 64, 1);         // [5,5,64] f16
    launch_conv(stream, d4buf, 64, 1, nullptr, 0, woutT, nullptr, ex_bout, nullptr,
                nullptr, out_f + 24, 4, 512, 512, 5, 1, 1, 0, 1, selbuf, 5L * 64, 5);
}
